// MultiHeadAttention_86397562126350
// MI455X (gfx1250) — compile-verified, matrix operand order checked by reading, not run on hardware
//
#include <hip/hip_runtime.h>
#include <hip/hip_bf16.h>
#include <math.h>

// ---------------------------------------------------------------------------
// MI455X (gfx1250) fused multi-head attention forward.
// Pipeline: cast->f16, fused QKV GEMM + RoPE, flash attention (online
// softmax), output projection.
//   - GEMMs:   v_wmma_f32_16x16x32_f16
//   - Weights: staged into LDS via TDM tensor_load_to_lds (TENSORcnt)
//   - P tiles: LDS transpose with explicit s_wait_dscnt
// ---------------------------------------------------------------------------

typedef __attribute__((ext_vector_type(16))) _Float16 v16h;
typedef __attribute__((ext_vector_type(8)))  _Float16 v8h;
typedef __attribute__((ext_vector_type(4)))  _Float16 v4h;
typedef __attribute__((ext_vector_type(8)))  float    v8f;
typedef __attribute__((ext_vector_type(4)))  unsigned int u32x4;
typedef __attribute__((ext_vector_type(8)))  unsigned int u32x8;

#define DMODEL   1024
#define NHEADS   16
#define HDIM     64
#define BATCH    4
#define SEQ      1024
#define NROWS    (BATCH*SEQ)          // 4096
#define KPANEL   256                  // k-columns staged per TDM transfer
#define LN10000  9.210340371976184f   // ln(10000)

// Build a 16-half A/B fragment from two contiguous 16-byte chunks.
// Per CDNA5 16-bit A layout: element e in [0,8): k = 8*laneHi + e
//                            element e in [8,16): k = 16 + 8*laneHi + (e-8)
static __device__ __forceinline__ v16h mk16(v8h lo, v8h hi) {
  v16h r;
#pragma unroll
  for (int j = 0; j < 8; ++j) { r[j] = lo[j]; r[j + 8] = hi[j]; }
  return r;
}

static __device__ __forceinline__ v16h ldFrag(const _Float16* row, int kbase, int lh) {
  v8h a0 = *(const v8h*)(row + kbase + lh * 8);
  v8h a1 = *(const v8h*)(row + kbase + 16 + lh * 8);
  return mk16(a0, a1);
}

// ---------------------------------------------------------------------------
// TDM: DMA a [rows x cols] f16 tile (row stride `stride` elems) from global
// memory into LDS at byte offset lds_off. One issue per wave; tracked with
// TENSORcnt. D# built per CDNA5 ISA (groups 0 and 1,
// 2D tensor, VADDR2/3 = NULL).
// ---------------------------------------------------------------------------
static __device__ __forceinline__ void tdm_load_tile_f16(
    const _Float16* gsrc, unsigned lds_off,
    unsigned rows, unsigned cols, unsigned stride) {
  unsigned long long ga = (unsigned long long)(uintptr_t)gsrc;
  u32x4 g0;
  g0[0] = 1u;                                   // count=1, user descriptor
  g0[1] = lds_off;                              // lds_addr (bytes)
  g0[2] = (unsigned)(ga & 0xffffffffu);         // global_addr[31:0]
  g0[3] = (unsigned)((ga >> 32) & 0x01ffffffu)  // global_addr[56:32]
          | 0x80000000u;                        // type=2 ("image")
  u32x8 g1;
  g1[0] = 1u << 16;                             // data_size=1 (2 bytes/elem)
  g1[1] = (cols & 0xffffu) << 16;               // tensor_dim0[15:0]
  g1[2] = (cols >> 16) | ((rows & 0xffffu) << 16);   // tdim0[31:16]|tdim1[15:0]
  g1[3] = (rows >> 16) | ((cols & 0xffffu) << 16);   // tdim1[31:16]|tile_dim0
  g1[4] = rows & 0xffffu;                       // tile_dim1 (tile_dim2 = 0)
  g1[5] = stride;                               // tensor_dim0_stride[31:0]
  g1[6] = 0u;                                   // t0stride[47:32] | t1stride[15:0]
  g1[7] = 0u;                                   // t1stride[47:16]
  asm volatile("tensor_load_to_lds %0, %1" :: "s"(g0), "s"(g1) : "memory");
}

// ---------------------------------------------------------------------------
// Kernel 0: f32 -> f16 cast (vectorized)
// ---------------------------------------------------------------------------
__global__ void cast_f32_to_f16(const float* __restrict__ src,
                                _Float16* __restrict__ dst, int n4) {
  int i = blockIdx.x * blockDim.x + threadIdx.x;
  int stride = gridDim.x * blockDim.x;
  const float4* s4 = (const float4*)src;
  v4h* d4 = (v4h*)dst;
  for (; i < n4; i += stride) {
    float4 f = s4[i];
    v4h h;
    h[0] = (_Float16)f.x; h[1] = (_Float16)f.y;
    h[2] = (_Float16)f.z; h[3] = (_Float16)f.w;
    d4[i] = h;
  }
}

// ---------------------------------------------------------------------------
// Kernel 1: fused QKV projection + RoPE.
//   grid = (64, 16, 3)  block = (32,4)
//   blockIdx.x*4+ty -> 16-row M tile of x; blockIdx.y -> head (64-col N tile)
//   blockIdx.z      -> 0:Q  1:K  2:V
//   Weight panel (64 x 256 f16 per step) staged via TDM into LDS, shared by
//   all 4 waves. Q,K stored [b,h,t,d] RoPE'd; V stored transposed [b,h,d,t].
// ---------------------------------------------------------------------------
__global__ __launch_bounds__(128) void qkv_rope_kernel(
    const _Float16* __restrict__ xh,
    const _Float16* __restrict__ Wqh,
    const _Float16* __restrict__ Wkh,
    const _Float16* __restrict__ Wvh,
    _Float16* __restrict__ Qh,
    _Float16* __restrict__ Kh,
    _Float16* __restrict__ Vth) {
  __shared__ __align__(16) _Float16 tileB[64 * KPANEL];   // 32 KB weight panel
  __shared__ __align__(16) _Float16 lbuf[4][16 * 64];     //  8 KB transpose buf

  const int lane = threadIdx.x;
  const int wv   = threadIdx.y;
  const int lh   = lane >> 4;
  const int nl   = lane & 15;
  const int z    = blockIdx.z;
  const int head = blockIdx.y;
  const int m_base = (blockIdx.x * 4 + wv) * 16;
  const int n_base = head * 64;
  const _Float16* W = (z == 0) ? Wqh : (z == 1) ? Wkh : Wvh;

  const _Float16* arow = xh + (size_t)(m_base + nl) * DMODEL;

  v8f acc[4];
#pragma unroll
  for (int s = 0; s < 4; ++s)
#pragma unroll
    for (int r = 0; r < 8; ++r) acc[s][r] = 0.0f;

  for (int kp = 0; kp < DMODEL; kp += KPANEL) {
    if (kp) __syncthreads();             // all waves done with previous panel
    if (wv == 0) {
      tdm_load_tile_f16(W + (size_t)n_base * DMODEL + kp,
                        (unsigned)(uintptr_t)tileB, 64u, KPANEL, DMODEL);
      __builtin_amdgcn_s_wait_tensorcnt(0);
    }
    __syncthreads();                     // panel visible to all waves

    for (int kk = 0; kk < KPANEL; kk += 32) {
      __builtin_prefetch(arow + kp + kk + 64, 0, 1);   // global_prefetch_b8
      v16h a = ldFrag(arow, kp + kk, lh);
#pragma unroll
      for (int s = 0; s < 4; ++s) {
        const _Float16* brow = tileB + (s * 16 + nl) * KPANEL;
        v16h b = ldFrag(brow, kk, lh);
        acc[s] = __builtin_amdgcn_wmma_f32_16x16x32_f16(
            false, a, false, b, (short)0, acc[s], false, false);
      }
    }
  }

  const int bb = m_base >> 10;        // batch index (tile never crosses batch)
  const int t0 = m_base & 1023;       // time offset of the tile
  const int bh = bb * NHEADS + head;

  if (z < 2) {
    // ---- RoPE in-register: adjacent head dims live in adjacent lanes ----
#pragma unroll
    for (int s = 0; s < 4; ++s) {
      int d = s * 16 + nl;
      int fi = d >> 1;
      float invf = __expf(-(float)fi * (LN10000 / 32.0f));
#pragma unroll
      for (int r = 0; r < 8; ++r) {
        int t = t0 + r + 8 * lh;
        float ang = (float)t * invf;
        float cs = __cosf(ang), sn = __sinf(ang);
        float v0 = acc[s][r];
        float pv = __shfl_xor(v0, 1, 32);   // rotary partner
        acc[s][r] = (d & 1) ? (pv * sn + v0 * cs) : (v0 * cs - pv * sn);
      }
    }
    // ---- LDS transpose: scatter b16 -> coalesced b128 global stores ----
    _Float16* lb = lbuf[wv];
#pragma unroll
    for (int s = 0; s < 4; ++s) {
      int d = s * 16 + nl;
#pragma unroll
      for (int r = 0; r < 8; ++r) {
        int m = r + 8 * lh;
        lb[m * 64 + d] = (_Float16)acc[s][r];
      }
    }
    asm volatile("s_wait_dscnt 0" ::: "memory");
    _Float16* dst = ((z == 0) ? Qh : Kh) + ((size_t)bh * SEQ + t0) * HDIM;
#pragma unroll
    for (int p = 0; p < 4; ++p) {
      int unit = p * 32 + lane;
      int m = unit >> 3, ch = unit & 7;
      v8h w8 = *(const v8h*)(lb + m * 64 + ch * 8);
      *(v8h*)(dst + (size_t)m * HDIM + ch * 8) = w8;
    }
  } else {
    // ---- V: C-layout packs directly into transposed [b,h,d,t] b128 stores
#pragma unroll
    for (int s = 0; s < 4; ++s) {
      int d = s * 16 + nl;
      v8h w8;
#pragma unroll
      for (int r = 0; r < 8; ++r) w8[r] = (_Float16)acc[s][r];
      *(v8h*)(Vth + ((size_t)bh * HDIM + d) * SEQ + t0 + 8 * lh) = w8;
    }
  }
}

// ---------------------------------------------------------------------------
// Kernel 2: flash attention with online softmax.
//   grid = (16, 64)  block = (32,4)
//   blockIdx.x -> 64-row q tile, blockIdx.y -> (b,h); wave owns 16 q rows.
// ---------------------------------------------------------------------------
__global__ __launch_bounds__(128) void flash_attn_kernel(
    const _Float16* __restrict__ Qh,
    const _Float16* __restrict__ Kh,
    const _Float16* __restrict__ Vth,
    _Float16* __restrict__ AOh) {
  __shared__ __align__(16) _Float16 pbuf[4][16 * 64];

  const int lane = threadIdx.x;
  const int wv   = threadIdx.y;
  const int lh   = lane >> 4;
  const int nl   = lane & 15;
  const int qtile = blockIdx.x;
  const int bh    = blockIdx.y;
  const int q0    = qtile * 64 + wv * 16;
  const float scale = 0.125f;   // 1/sqrt(64)

  // Cache this wave's 16x64 Q tile in registers (A-operand fragments).
  const _Float16* qrow = Qh + ((size_t)bh * SEQ + q0 + nl) * HDIM;
  v16h aq0 = ldFrag(qrow, 0, lh);
  v16h aq1 = ldFrag(qrow, 32, lh);

  float mrun[8], lsum[8];
  v8f accO[4];
#pragma unroll
  for (int r = 0; r < 8; ++r) { mrun[r] = -1e30f; lsum[r] = 0.0f; }
#pragma unroll
  for (int s = 0; s < 4; ++s)
#pragma unroll
    for (int r = 0; r < 8; ++r) accO[s][r] = 0.0f;

  _Float16* lb = pbuf[wv];
  const int nkb = qtile + 1;   // causal: key blocks 0..qtile

  for (int kb = 0; kb < nkb; ++kb) {
    const int j0 = kb * 64;

    // ---- S = (Q Kᵀ) * scale, 16x64 per wave ----
    v8f sc[4];
#pragma unroll
    for (int s = 0; s < 4; ++s)
#pragma unroll
      for (int r = 0; r < 8; ++r) sc[s][r] = 0.0f;

#pragma unroll
    for (int it = 0; it < 2; ++it) {
      v16h a = (it == 0) ? aq0 : aq1;
#pragma unroll
      for (int s = 0; s < 4; ++s) {
        const _Float16* krow =
            Kh + ((size_t)bh * SEQ + j0 + s * 16 + nl) * HDIM;
        v16h bk = ldFrag(krow, it * 32, lh);
        sc[s] = __builtin_amdgcn_wmma_f32_16x16x32_f16(
            false, a, false, bk, (short)0, sc[s], false, false);
      }
    }

    // ---- scale + causal mask ----
#pragma unroll
    for (int s = 0; s < 4; ++s) {
      int j = j0 + s * 16 + nl;
#pragma unroll
      for (int r = 0; r < 8; ++r) {
        int tq = q0 + r + 8 * lh;
        float v0 = sc[s][r] * scale;
        sc[s][r] = (j > tq) ? -1e30f : v0;
      }
    }

    // ---- online softmax: per-row stats via 16-lane shfl_xor reductions ----
#pragma unroll
    for (int r = 0; r < 8; ++r) {
      float rv = fmaxf(fmaxf(sc[0][r], sc[1][r]), fmaxf(sc[2][r], sc[3][r]));
#pragma unroll
      for (int off = 1; off < 16; off <<= 1)
        rv = fmaxf(rv, __shfl_xor(rv, off, 32));
      float mnew = fmaxf(mrun[r], rv);
      float corr = __expf(mrun[r] - mnew);
      float rs = 0.0f;
#pragma unroll
      for (int s = 0; s < 4; ++s) {
        float p = __expf(sc[s][r] - mnew);
        sc[s][r] = p;
        rs += p;
      }
#pragma unroll
      for (int off = 1; off < 16; off <<= 1)
        rs += __shfl_xor(rs, off, 32);
      lsum[r] = lsum[r] * corr + rs;
      mrun[r] = mnew;
#pragma unroll
      for (int s = 0; s < 4; ++s) accO[s][r] *= corr;
    }

    // ---- P (C-layout) -> LDS row-major [m][64] for A-operand reload ----
#pragma unroll
    for (int s = 0; s < 4; ++s) {
      int jl = s * 16 + nl;
#pragma unroll
      for (int r = 0; r < 8; ++r) {
        int m = r + 8 * lh;
        lb[m * 64 + jl] = (_Float16)sc[s][r];
      }
    }
    asm volatile("s_wait_dscnt 0" ::: "memory");

    // ---- O += P · V  (V stored transposed = native B layout) ----
#pragma unroll
    for (int it = 0; it < 2; ++it) {
      v16h ap = ldFrag(lb + nl * 64, it * 32, lh);
#pragma unroll
      for (int s = 0; s < 4; ++s) {
        const _Float16* vrow =
            Vth + ((size_t)bh * HDIM + s * 16 + nl) * SEQ + j0;
        v16h bv = ldFrag(vrow, it * 32, lh);
        accO[s] = __builtin_amdgcn_wmma_f32_16x16x32_f16(
            false, ap, false, bv, (short)0, accO[s], false, false);
      }
    }
    asm volatile("s_wait_dscnt 0" ::: "memory");
  }

  // ---- epilogue: normalize, write attn-out [B,T,C] via LDS transpose ----
  const int bb = bh >> 4, hh = bh & 15;
#pragma unroll
  for (int s = 0; s < 4; ++s) {
    int d = s * 16 + nl;
#pragma unroll
    for (int r = 0; r < 8; ++r) {
      int m = r + 8 * lh;
      lb[m * 64 + d] = (_Float16)(accO[s][r] / lsum[r]);
    }
  }
  asm volatile("s_wait_dscnt 0" ::: "memory");
  _Float16* dst = AOh + ((size_t)bb * SEQ + q0) * DMODEL + hh * HDIM;
#pragma unroll
  for (int p = 0; p < 4; ++p) {
    int unit = p * 32 + lane;
    int m = unit >> 3, ch = unit & 7;
    v8h w8 = *(const v8h*)(lb + m * 64 + ch * 8);
    *(v8h*)(dst + (size_t)m * DMODEL + ch * 8) = w8;
  }
}

// ---------------------------------------------------------------------------
// Kernel 3: output projection  out = AO @ Woᵀ  (f32 result, coalesced)
//   grid = (64, 16)  block = (32,4); Wo panels staged via TDM like kernel 1.
// ---------------------------------------------------------------------------
__global__ __launch_bounds__(128) void out_proj_kernel(
    const _Float16* __restrict__ AOh,
    const _Float16* __restrict__ Woh,
    float* __restrict__ out) {
  __shared__ __align__(16) _Float16 tileB[64 * KPANEL];   // 32 KB weight panel
  __shared__ __align__(16) float obuf[4][16 * 64];        // 16 KB

  const int lane = threadIdx.x;
  const int wv   = threadIdx.y;
  const int lh   = lane >> 4;
  const int nl   = lane & 15;
  const int m_base = (blockIdx.x * 4 + wv) * 16;
  const int n_base = blockIdx.y * 64;

  const _Float16* arow = AOh + (size_t)(m_base + nl) * DMODEL;

  v8f acc[4];
#pragma unroll
  for (int s = 0; s < 4; ++s)
#pragma unroll
    for (int r = 0; r < 8; ++r) acc[s][r] = 0.0f;

  for (int kp = 0; kp < DMODEL; kp += KPANEL) {
    if (kp) __syncthreads();
    if (wv == 0) {
      tdm_load_tile_f16(Woh + (size_t)n_base * DMODEL + kp,
                        (unsigned)(uintptr_t)tileB, 64u, KPANEL, DMODEL);
      __builtin_amdgcn_s_wait_tensorcnt(0);
    }
    __syncthreads();

    for (int kk = 0; kk < KPANEL; kk += 32) {
      v16h a = ldFrag(arow, kp + kk, lh);
#pragma unroll
      for (int s = 0; s < 4; ++s) {
        const _Float16* brow = tileB + (s * 16 + nl) * KPANEL;
        v16h b = ldFrag(brow, kk, lh);
        acc[s] = __builtin_amdgcn_wmma_f32_16x16x32_f16(
            false, a, false, b, (short)0, acc[s], false, false);
      }
    }
  }

  float* ob = obuf[wv];
#pragma unroll
  for (int s = 0; s < 4; ++s) {
    int n = s * 16 + nl;
#pragma unroll
    for (int r = 0; r < 8; ++r) {
      int m = r + 8 * lh;
      ob[m * 64 + n] = acc[s][r];
    }
  }
  asm volatile("s_wait_dscnt 0" ::: "memory");
#pragma unroll
  for (int p = 0; p < 8; ++p) {
    int unit = p * 32 + lane;
    int m = unit >> 4, ch = unit & 15;
    float4 f = *(const float4*)(ob + m * 64 + ch * 4);
    *(float4*)(out + (size_t)(m_base + m) * DMODEL + n_base + ch * 4) = f;
  }
}

// ---------------------------------------------------------------------------
// Host launcher
// ---------------------------------------------------------------------------
extern "C" void kernel_launch(void* const* d_in, const int* in_sizes, int n_in,
                              void* d_out, int out_size, void* d_ws,
                              size_t ws_size, hipStream_t stream) {
  const float* x  = (const float*)d_in[0];
  const float* Wq = (const float*)d_in[1];
  const float* Wk = (const float*)d_in[2];
  const float* Wv = (const float*)d_in[3];
  const float* Wo = (const float*)d_in[4];
  float* out = (float*)d_out;

  char* ws = (char*)d_ws;
  const size_t MB = (size_t)1 << 20;
  _Float16* xh  = (_Float16*)(ws + 0);        //  8 MB  x as f16
  _Float16* Wqh = (_Float16*)(ws + 8 * MB);   //  2 MB
  _Float16* Wkh = (_Float16*)(ws + 10 * MB);  //  2 MB
  _Float16* Wvh = (_Float16*)(ws + 12 * MB);  //  2 MB
  _Float16* Woh = (_Float16*)(ws + 14 * MB);  //  2 MB
  _Float16* Qh  = (_Float16*)(ws + 16 * MB);  //  8 MB  [b,h,t,d]
  _Float16* Kh  = (_Float16*)(ws + 24 * MB);  //  8 MB  [b,h,t,d]
  _Float16* Vth = (_Float16*)(ws + 32 * MB);  //  8 MB  [b,h,d,t]
  _Float16* AOh = (_Float16*)(ws + 40 * MB);  //  8 MB  [b,t,c]

  cast_f32_to_f16<<<1024, 256, 0, stream>>>(x,  xh,  (NROWS * DMODEL) / 4);
  cast_f32_to_f16<<<256,  256, 0, stream>>>(Wq, Wqh, (DMODEL * DMODEL) / 4);
  cast_f32_to_f16<<<256,  256, 0, stream>>>(Wk, Wkh, (DMODEL * DMODEL) / 4);
  cast_f32_to_f16<<<256,  256, 0, stream>>>(Wv, Wvh, (DMODEL * DMODEL) / 4);
  cast_f32_to_f16<<<256,  256, 0, stream>>>(Wo, Woh, (DMODEL * DMODEL) / 4);

  qkv_rope_kernel<<<dim3(64, 16, 3), dim3(32, 4, 1), 0, stream>>>(
      xh, Wqh, Wkh, Wvh, Qh, Kh, Vth);

  flash_attn_kernel<<<dim3(16, 64, 1), dim3(32, 4, 1), 0, stream>>>(
      Qh, Kh, Vth, AOh);

  out_proj_kernel<<<dim3(64, 16, 1), dim3(32, 4, 1), 0, stream>>>(
      AOh, Woh, out);
}
